// ModulatedDeformConv2d_69303592288915
// MI455X (gfx1250) — matrix operand order repaired and checked by reading, not compiled, not run on hardware
//
#include <hip/hip_runtime.h>
#include <hip/hip_bf16.h>

typedef __attribute__((ext_vector_type(16))) __bf16 v16bf;
typedef __attribute__((ext_vector_type(8)))  __bf16 v8bf;
typedef __attribute__((ext_vector_type(8)))  float  v8f;

#define B_     4
#define CIN_   128
#define H_     128
#define W_     128
#define COUT_  128
#define KK_    9
#define DG_    8
#define CG_    16
#define HO_    128
#define WO_    128
#define HW_    (H_ * W_)
#define KDIM_  (CIN_ * KK_)        // 1152
#define NT_    32                  // output positions per workgroup
#define KSTEPS_ (KDIM_ / 32)       // 36 WMMA K-steps
#define CROW_  (KDIM_ + 8)         // padded cols row (halves): bank-conflict-free B loads
#define PRM_N  (DG_ * KK_ * NT_)   // 2304 bilinear parameter sets
#define SMEM_BYTES (PRM_N * 32 + NT_ * CROW_ * 2)   // 73728 + 74240 = 147968 B

struct Prm { int idx[4]; float w[4]; };             // 32 B

// ---- convert weight fp32 (COUT, CIN*KK) -> bf16 [O][K] (same linear layout) ----
__global__ void wprep_kernel(const float* __restrict__ w, __bf16* __restrict__ wt) {
    int i = blockIdx.x * 256 + threadIdx.x;
    if (i < COUT_ * KDIM_) wt[i] = (__bf16)w[i];
}

__global__ __launch_bounds__(256)
void dcn_fused_kernel(const float* __restrict__ x,
                      const float* __restrict__ offs,
                      const float* __restrict__ mask,
                      const __bf16* __restrict__ wt,
                      const float* __restrict__ bias,
                      float* __restrict__ out) {
    extern __shared__ char smem[];
    Prm*    prm  = (Prm*)smem;
    __bf16* cols = (__bf16*)(smem + PRM_N * sizeof(Prm));

    const int tid = threadIdx.x;
    const int g0  = blockIdx.x * NT_;       // first linear output position of tile
    const int b   = g0 >> 14;               // HO*WO = 16384, NT divides it -> uniform b
    const int posbase = g0 & 16383;

    // ================= Stage A: bilinear params per (dg, kk, n) =================
    #pragma unroll
    for (int it = 0; it < PRM_N / 256; ++it) {      // 9 iterations
        int e  = it * 256 + tid;
        int n  = e & (NT_ - 1);
        int t  = e / NT_;                   // dg*KK + kk
        int kk = t % KK_;
        int dg = t / KK_;
        int pos = posbase + n;
        int ho = pos >> 7, wo = pos & 127;

        int och = (dg * KK_ + kk) * 2;      // offset reshaped (B, DG, KK, 2, HO, WO)
        size_t obase = ((size_t)b * (DG_ * 2 * KK_) + och) * HW_ + (size_t)ho * WO_ + wo;
        float oy = offs[obase];
        float ox = offs[obase + HW_];
        float mv = mask[((size_t)b * (DG_ * KK_) + dg * KK_ + kk) * HW_ + (size_t)ho * WO_ + wo];

        int ky = kk / 3, kx = kk % 3;
        float ys = (float)(ho - 1 + ky) + oy;       // stride=1, pad=1, dil=1
        float xs = (float)(wo - 1 + kx) + ox;
        float y0f = floorf(ys), x0f = floorf(xs);
        float ly = ys - y0f, lx = xs - x0f;
        int y0 = (int)y0f, x0 = (int)x0f;
        int y1 = y0 + 1,   x1 = x0 + 1;
        float vy0 = (y0 >= 0 && y0 < H_) ? 1.0f : 0.0f;
        float vy1 = (y1 >= 0 && y1 < H_) ? 1.0f : 0.0f;
        float vx0 = (x0 >= 0 && x0 < W_) ? 1.0f : 0.0f;
        float vx1 = (x1 >= 0 && x1 < W_) ? 1.0f : 0.0f;
        int y0c = min(max(y0, 0), H_ - 1), y1c = min(max(y1, 0), H_ - 1);
        int x0c = min(max(x0, 0), W_ - 1), x1c = min(max(x1, 0), W_ - 1);

        Prm q;
        q.idx[0] = y0c * W_ + x0c;  q.w[0] = (1.0f - ly) * (1.0f - lx) * vy0 * vx0 * mv;
        q.idx[1] = y0c * W_ + x1c;  q.w[1] = (1.0f - ly) * lx          * vy0 * vx1 * mv;
        q.idx[2] = y1c * W_ + x0c;  q.w[2] = ly          * (1.0f - lx) * vy1 * vx0 * mv;
        q.idx[3] = y1c * W_ + x1c;  q.w[3] = ly          * lx          * vy1 * vx1 * mv;
        prm[e] = q;
    }
    __syncthreads();

    // ================= Stage B: build cols tile [n][K] bf16 in LDS =================
    #pragma unroll
    for (int it = 0; it < (CIN_ * NT_) / 256; ++it) {   // 16 iterations
        int p = it * 256 + tid;
        int n = p & (NT_ - 1);                          // tid-fast -> coalesced gathers
        int c = p / NT_;
        int dg = c >> 4;                                // CG = 16
        const float* xb = x + ((size_t)b * CIN_ + c) * HW_;
        __bf16* crow = cols + (size_t)n * CROW_ + c * KK_;
        const Prm* pp = prm + (dg * KK_) * NT_ + n;
        #pragma unroll
        for (int kk = 0; kk < KK_; ++kk) {
            Prm q = pp[kk * NT_];
            float v = q.w[0] * xb[q.idx[0]] + q.w[1] * xb[q.idx[1]]
                    + q.w[2] * xb[q.idx[2]] + q.w[3] * xb[q.idx[3]];
            crow[kk] = (__bf16)v;
        }
    }
    __syncthreads();

    // ================= Stage C: WMMA GEMM, 8 waves x 2 (16x16) tiles =================
    const int lane = tid & 31;
    const int wv   = tid >> 5;
    const int obase = wv * 16;              // M tile
    const int lrow = lane & 15;
    const bool hiHalf = lane >= 16;

    // A per-lane base: row = obase+lrow; lanes<16 read K 0..15, lanes>=16 K 16..31
    const __bf16* arow  = wt + (size_t)(obase + lrow) * KDIM_ + (hiHalf ? 16 : 0);
    // B per-lane base: column n = lrow (+16 for second tile); lanes<16 K 0..15, lanes>=16 K 16..31
    const __bf16* brow0 = cols + (size_t)lrow * CROW_ + (hiHalf ? 16 : 0);
    const __bf16* brow1 = brow0 + 16 * CROW_;

    v8f acc0 = {}; v8f acc1 = {};
    for (int ks = 0; ks < KSTEPS_; ++ks) {
        const __bf16* ap  = arow  + ks * 32;
        const __bf16* bp0 = brow0 + ks * 32;
        const __bf16* bp1 = brow1 + ks * 32;
        if (ks + 1 < KSTEPS_) __builtin_prefetch(arow + (ks + 1) * 32, 0, 3);

        v16bf a, b0, b1;
        ((v8bf*)&a)[0]  = *(const v8bf*)ap;          // K 0..7  (or 16..23)
        ((v8bf*)&a)[1]  = *(const v8bf*)(ap + 8);    // K 8..15 (or 24..31)
        ((v8bf*)&b0)[0] = *(const v8bf*)bp0;         // K 0..7  (or 16..23)
        ((v8bf*)&b0)[1] = *(const v8bf*)(bp0 + 8);   // K 8..15 (or 24..31)
        ((v8bf*)&b1)[0] = *(const v8bf*)bp1;
        ((v8bf*)&b1)[1] = *(const v8bf*)(bp1 + 8);

        acc0 = __builtin_amdgcn_wmma_f32_16x16x32_bf16(false, a, false, b0,
                                                       (short)0, acc0, false, false);
        acc1 = __builtin_amdgcn_wmma_f32_16x16x32_bf16(false, a, false, b1,
                                                       (short)0, acc1, false, false);
    }

    // ---- epilogue: bias + store. acc VGPR r: M = r (lanes<16) / r+8 (lanes>=16), N = lane&15
    float* obb = out + (size_t)b * COUT_ * (size_t)HW_ + posbase;
    #pragma unroll
    for (int r = 0; r < 8; ++r) {
        int o = obase + r + (hiHalf ? 8 : 0);
        float bv = bias[o];
        float* op = obb + (size_t)o * HW_;
        op[lrow]      = acc0[r] + bv;
        op[lrow + 16] = acc1[r] + bv;
    }
}

extern "C" void kernel_launch(void* const* d_in, const int* in_sizes, int n_in,
                              void* d_out, int out_size, void* d_ws, size_t ws_size,
                              hipStream_t stream) {
    const float* x    = (const float*)d_in[0];
    const float* offs = (const float*)d_in[1];
    const float* mask = (const float*)d_in[2];
    const float* wgt  = (const float*)d_in[3];
    const float* bias = (const float*)d_in[4];
    __bf16* wt = (__bf16*)d_ws;                  // 294,912 B of scratch

    (void)in_sizes; (void)n_in; (void)out_size; (void)ws_size;

    hipFuncSetAttribute(reinterpret_cast<const void*>(dcn_fused_kernel),
                        hipFuncAttributeMaxDynamicSharedMemorySize, SMEM_BYTES);

    wprep_kernel<<<(COUT_ * KDIM_ + 255) / 256, 256, 0, stream>>>(wgt, wt);
    dcn_fused_kernel<<<(B_ * HO_ * WO_) / NT_, 256, SMEM_BYTES, stream>>>(
        x, offs, mask, wt, bias, (float*)d_out);
}
